// KSpaceModule_50122268344878
// MI455X (gfx1250) — matrix operand order repaired and checked by reading, not compiled, not run on hardware
//
#include <hip/hip_runtime.h>
#include <hip/hip_bf16.h>

typedef __attribute__((ext_vector_type(16))) _Float16 v16h;
typedef __attribute__((ext_vector_type(8)))  float    v8f;

#define NBINS        64
#define THREADS      256
#define SCAT_BLOCKS  1024
#define GATH_BLOCKS  1024

// Branch-free tanh: prefer CDNA5 hardware v_tanh_f32; else exp2/rcp fallback.
__device__ __forceinline__ float fast_tanh(float x) {
#if __has_builtin(__builtin_amdgcn_tanhf)
  return __builtin_amdgcn_tanhf(x);
#else
  float e = __builtin_amdgcn_exp2f(x * 2.88539008177792681472f);  // e^(2x)
  float r = __builtin_amdgcn_rcpf(e + 1.0f);
  return 1.0f - 2.0f * r;   // saturates correctly to +/-1 for large |x|
#endif
}

__device__ __forceinline__ void inv3x3(const float* a, float* ic) {
  float det = a[0]*(a[4]*a[8]-a[5]*a[7]) - a[1]*(a[3]*a[8]-a[5]*a[6])
            + a[2]*(a[3]*a[7]-a[4]*a[6]);
  float id = 1.0f/det;
  ic[0]=(a[4]*a[8]-a[5]*a[7])*id; ic[1]=(a[2]*a[7]-a[1]*a[8])*id; ic[2]=(a[1]*a[5]-a[2]*a[4])*id;
  ic[3]=(a[5]*a[6]-a[3]*a[8])*id; ic[4]=(a[0]*a[8]-a[2]*a[6])*id; ic[5]=(a[2]*a[3]-a[0]*a[5])*id;
  ic[6]=(a[3]*a[7]-a[4]*a[6])*id; ic[7]=(a[1]*a[6]-a[0]*a[7])*id; ic[8]=(a[0]*a[4]-a[1]*a[3])*id;
}

// TSC order-3 weights + wrapped mesh indices (NS=4), rel_j = (pos . inv_cell)_j * 4
__device__ __forceinline__ void p3m_wi(float px, float py, float pz, const float* ic,
                                       float w[3][3], int ix[3][3]) {
  float rel[3];
  rel[0] = (px*ic[0]+py*ic[3]+pz*ic[6])*4.0f;
  rel[1] = (px*ic[1]+py*ic[4]+pz*ic[7])*4.0f;
  rel[2] = (px*ic[2]+py*ic[5]+pz*ic[8])*4.0f;
  #pragma unroll
  for (int d = 0; d < 3; ++d) {
    float i0 = floorf(rel[d] + 0.5f);
    float x  = rel[d] - i0;
    float x2 = x*x;
    w[0][d] = 0.125f*(1.f - 4.f*x + 4.f*x2);
    w[1][d] = 0.75f - x2;
    w[2][d] = 0.125f*(1.f + 4.f*x + 4.f*x2);
    int ib = (int)i0;
    ix[0][d] = (ib-1)&3; ix[1][d] = ib&3; ix[2][d] = (ib+1)&3;
  }
}

// ---- Kernel 1: scatter charges onto per-block LDS mesh, emit per-block partials ----
__global__ void __launch_bounds__(THREADS)
scatter_kernel(const float* __restrict__ pos, const float* __restrict__ cell,
               const float* __restrict__ q, int N, float* __restrict__ partials) {
  __shared__ float mesh[NBINS];
  if (threadIdx.x < NBINS) mesh[threadIdx.x] = 0.f;
  __syncthreads();
  float ic[9]; inv3x3(cell, ic);
  int stride = gridDim.x * blockDim.x;
  for (int i = blockIdx.x*blockDim.x + threadIdx.x; i < N; i += stride) {
    float px = pos[3*i], py = pos[3*i+1], pz = pos[3*i+2];
    float qq = q[i];
    float w[3][3]; int ix[3][3];
    p3m_wi(px, py, pz, ic, w, ix);
    #pragma unroll
    for (int a = 0; a < 3; ++a) {
      float wa = qq * w[a][0]; int ba = ix[a][0]*16;
      #pragma unroll
      for (int b = 0; b < 3; ++b) {
        float wab = wa * w[b][1]; int bb = ba + ix[b][1]*4;
        #pragma unroll
        for (int c = 0; c < 3; ++c)
          atomicAdd(&mesh[bb + ix[c][2]], wab * w[c][2]);   // ds_add_f32
      }
    }
  }
  __syncthreads();
  if (threadIdx.x < NBINS)
    partials[blockIdx.x*NBINS + threadIdx.x] = mesh[threadIdx.x];
}

// ---- Kernel 2: reduce partials; apply k-space filter as exact real circulant ----
// g[d] = (1/64) sum_n F(k(n)) * cos(2*pi*(n.d)/4); cos table is {1,0,-1,0}.
__global__ void filter_kernel(const float* __restrict__ partials, int nb,
                              const float* __restrict__ cell,
                              const float* __restrict__ sig2p,
                              float* __restrict__ fmout) {
  __shared__ float mesh[NBINS];
  __shared__ float g[NBINS];
  int t = threadIdx.x;                       // 64 threads, one per bin
  float s = 0.f;
  for (int b = 0; b < nb; ++b) s += partials[b*NBINS + t];
  mesh[t] = s;
  float ic[9]; inv3x3(cell, ic);
  const float TWO_PI = 6.28318530717958647692f;
  const float ctab[4] = {1.f, 0.f, -1.f, 0.f};
  const float ftab[4] = {0.f, 1.f, -2.f, -1.f};  // fftfreq(4)*4
  int dx = t>>4, dy = (t>>2)&3, dz = t&3;
  float sigma2 = sig2p[0];
  float acc = 0.f;
  for (int n = 1; n < 64; ++n) {            // n=0 term zeroed by reference
    int nx = n>>4, ny = (n>>2)&3, nz = n&3;
    float fx = ftab[nx], fy = ftab[ny], fz = ftab[nz];
    float kx = TWO_PI*(fx*ic[0] + fy*ic[1] + fz*ic[2]);
    float ky = TWO_PI*(fx*ic[3] + fy*ic[4] + fz*ic[5]);
    float kz = TWO_PI*(fx*ic[6] + fy*ic[7] + fz*ic[8]);
    float k2 = kx*kx + ky*ky + kz*kz;
    float F  = expf(-0.5f*sigma2*k2) / k2;
    acc += F * ctab[(nx*dx + ny*dy + nz*dz) & 3];
  }
  g[t] = acc * (1.0f/64.0f);
  __syncthreads();
  float fo = 0.f;
  for (int m = 0; m < 64; ++m) {
    int mx = m>>4, my = (m>>2)&3, mz = m&3;
    fo += g[((dx-mx)&3)*16 + ((dy-my)&3)*4 + ((dz-mz)&3)] * mesh[m];
  }
  fmout[t] = fo;
}

// ---- Kernel 3: gather potential + WMMA MLP + block reduction ----
// Per 16-particle tile: C1 = X(16xK32,f16) * W1^T ; C2 = tanh(C1+b1) * W2^T ;
// per-lane dot with W3 over the C2 fragment accumulates the global sum.
__global__ void __launch_bounds__(THREADS)
gather_mlp_kernel(const float* __restrict__ pos, const float* __restrict__ cell,
                  const float* __restrict__ q, int N, int nchunks,
                  const float* __restrict__ fmesh,
                  const float* __restrict__ W1, const float* __restrict__ b1,
                  const float* __restrict__ W2, const float* __restrict__ b2,
                  const float* __restrict__ W3,
                  float* __restrict__ partials) {
  __shared__ float    fm[NBINS];
  __shared__ _Float16 xs[8][64];       // per-wave: 32 particles x (q,pot)
  __shared__ _Float16 h1s[8][512];     // per-wave, double-buffered per tile: 2 x 16x16
  __shared__ float    red[THREADS];
  int tid  = threadIdx.x;
  int lane = tid & 31;
  int wid  = tid >> 5;
  int col  = lane & 15;
  bool lo  = lane < 16;
  bool act = col < 10;
  if (tid < NBINS) fm[tid] = fmesh[tid];

  // B fragments (cols striped over lanes 0..15; K=0..15 packed per lane; lanes 16..31 = K16..31 -> 0)
  v16h B1 = {};
  v16h B2 = {};
  if (lo && act) {
    B1[0] = (_Float16)W1[col*2 + 0];
    B1[1] = (_Float16)W1[col*2 + 1];
    #pragma unroll
    for (int k = 0; k < 10; ++k) B2[k] = (_Float16)W2[col*10 + k];
  }
  float b1l = act ? b1[col] : 0.f;
  float b2l = act ? b2[col] : 0.f;
  float w3l = act ? W3[col] : 0.f;
  float ic[9]; inv3x3(cell, ic);
  __syncthreads();

  float s = 0.f;
  for (int ch = blockIdx.x; ch < nchunks; ch += gridDim.x) {
    int i = ch*THREADS + tid;
    float qq = 0.f, pot = 0.f;
    if (i < N) {
      float px = pos[3*i], py = pos[3*i+1], pz = pos[3*i+2];
      qq = q[i];
      float w[3][3]; int ix[3][3];
      p3m_wi(px, py, pz, ic, w, ix);
      #pragma unroll
      for (int a = 0; a < 3; ++a) {
        float wa = w[a][0]; int ba = ix[a][0]*16;
        #pragma unroll
        for (int b = 0; b < 3; ++b) {
          float wab = wa * w[b][1]; int bb = ba + ix[b][1]*4;
          #pragma unroll
          for (int c = 0; c < 3; ++c) pot += wab * w[c][2] * fm[bb + ix[c][2]];
        }
      }
    }
    xs[wid][lane*2 + 0] = (_Float16)qq;
    xs[wid][lane*2 + 1] = (_Float16)pot;
    __syncthreads();                      // xs visible to whole wave (and block)
    #pragma unroll
    for (int t2 = 0; t2 < 2; ++t2) {      // two 16-particle tiles per wave
      // A1: row M = lane (lanes 0-15), K=0,1 = (q,pot); all other K zero.
      v16h A = {};
      if (lo) {
        A[0] = xs[wid][(t2*16 + lane)*2 + 0];
        A[1] = xs[wid][(t2*16 + lane)*2 + 1];
      }
      v8f c1 = {};
      c1 = __builtin_amdgcn_wmma_f32_16x16x32_f16(false, A, false, B1,
                                                  (short)0, c1, false, false);
      // tanh + bias, stage transposed through LDS (per-tile buffer) for next A
      _Float16* h1t = &h1s[wid][t2*256];
      int rowbase = lo ? 0 : 8;
      #pragma unroll
      for (int r = 0; r < 8; ++r)
        h1t[(rowbase + r)*16 + col] = (_Float16)fast_tanh(c1[r] + b1l);
      __syncthreads();                    // h1 tile visible
      v16h A2 = {};
      {
        // A layout: lanes 0-15 hold K0..15 (contiguous), lanes 16-31 hold K16..31 -> 0
        #pragma unroll
        for (int j = 0; j < 16; ++j) A2[j] = lo ? h1t[col*16 + j] : (_Float16)0;
      }
      v8f c2 = {};
      c2 = __builtin_amdgcn_wmma_f32_16x16x32_f16(false, A2, false, B2,
                                                  (short)0, c2, false, false);
      #pragma unroll
      for (int r = 0; r < 8; ++r) s += fast_tanh(c2[r] + b2l) * w3l;
    }
    // next iteration's xs store is ordered by its own __syncthreads; h1s is
    // double-buffered per tile, so no extra WAR barrier is needed here.
  }
  red[tid] = s;
  __syncthreads();
  for (int off = THREADS/2; off > 0; off >>= 1) {
    if (tid < off) red[tid] += red[tid + off];
    __syncthreads();
  }
  if (tid == 0) partials[blockIdx.x] = red[0];
}

// ---- Kernel 4: deterministic final sum; add N*b3, subtract padding's MLP(0) ----
__global__ void final_kernel(const float* __restrict__ partials, int nb,
                             const float* __restrict__ b1, const float* __restrict__ W2,
                             const float* __restrict__ b2, const float* __restrict__ W3,
                             const float* __restrict__ b3,
                             int N, int npad, float* __restrict__ out) {
  __shared__ float red[256];
  int tid = threadIdx.x;
  float s = 0.f;
  for (int i = tid; i < nb; i += 256) s += partials[i];
  red[tid] = s;
  __syncthreads();
  for (int off = 128; off > 0; off >>= 1) {
    if (tid < off) red[tid] += red[tid + off];
    __syncthreads();
  }
  if (tid == 0) {
    float h1[10], h2[10];
    for (int k = 0; k < 10; ++k) h1[k] = fast_tanh(b1[k]);
    float y0 = 0.f;
    for (int n = 0; n < 10; ++n) {
      float a = b2[n];
      for (int k = 0; k < 10; ++k) a += W2[n*10 + k] * h1[k];
      h2[n] = fast_tanh(a);
      y0 += W3[n] * h2[n];
    }
    out[0] = red[0] + (float)N * b3[0] - (float)npad * y0;
  }
}

extern "C" void kernel_launch(void* const* d_in, const int* in_sizes, int n_in,
                              void* d_out, int out_size, void* d_ws, size_t ws_size,
                              hipStream_t stream) {
  const float* pos  = (const float*)d_in[0];
  const float* cell = (const float*)d_in[1];
  const float* q    = (const float*)d_in[2];
  const float* sig2 = (const float*)d_in[3];
  const float* W1   = (const float*)d_in[4];
  const float* b1   = (const float*)d_in[5];
  const float* W2   = (const float*)d_in[6];
  const float* b2   = (const float*)d_in[7];
  const float* W3   = (const float*)d_in[8];
  const float* b3   = (const float*)d_in[9];
  int N = in_sizes[0] / 3;

  float* ws    = (float*)d_ws;
  float* scatP = ws;                          // SCAT_BLOCKS * 64 floats
  float* fmesh = ws + SCAT_BLOCKS*NBINS;      // 64 floats
  float* gathP = fmesh + NBINS;               // GATH_BLOCKS floats

  int nchunks = (N + THREADS - 1) / THREADS;
  int npad    = nchunks*THREADS - N;

  scatter_kernel<<<SCAT_BLOCKS, THREADS, 0, stream>>>(pos, cell, q, N, scatP);
  filter_kernel<<<1, 64, 0, stream>>>(scatP, SCAT_BLOCKS, cell, sig2, fmesh);
  gather_mlp_kernel<<<GATH_BLOCKS, THREADS, 0, stream>>>(pos, cell, q, N, nchunks, fmesh,
                                                         W1, b1, W2, b2, W3, gathP);
  final_kernel<<<1, 256, 0, stream>>>(gathP, GATH_BLOCKS, b1, W2, b2, W3, b3, N, npad,
                                      (float*)d_out);
}
